// VPNNetwork_21294447853603
// MI455X (gfx1250) — matrix operand order repaired and checked by reading, not compiled, not run on hardware
//
#include <hip/hip_runtime.h>
#include <hip/hip_bf16.h>
#include <cstdint>

// ---------------------------------------------------------------------------
// Problem constants (from the reference)
// ---------------------------------------------------------------------------
#define HH     64
#define WWID   64
#define KITER  64
#define BB     64
#define NFEAT  12288          // H*W*3
#define HW     4096           // H*W
#define PS     66             // padded stride for the 64x64 VIN grids

// WMMA vector types (gfx1250, wave32)
typedef __attribute__((ext_vector_type(16))) __bf16 v16bf;
typedef __attribute__((ext_vector_type(8)))  float  v8f;

// fp32 -> bf16: use the native cast; gfx1250 has hardware bf16 converts
// (lowered by clang to the packed cvt path, RNE).
__device__ __forceinline__ __bf16 f2bf(float f) { return (__bf16)f; }

// ---------------------------------------------------------------------------
// Kernel 1: phi = obs_flat @ phi_w + phi_b          [64,12288]x[12288,12288]
// Bandwidth-bound on phi_w (604 MB fp32 -> ~26us at 23.3 TB/s). bf16 WMMA,
// f32 accumulate keeps the compute requirement (~93 TF bf16) far below the
// matrix-core ceiling, so HBM stays the bound.
// One block = 8 waves = 64(M) x 64(N) tile; each wave owns a 16x32 strip
// (two accumulators sharing one A fragment). grid.x = 12288/64 = 192.
// ---------------------------------------------------------------------------
__global__ __launch_bounds__(256) void phi_gemm_kernel(
    const float* __restrict__ A,    // [64, 12288] row-major
    const float* __restrict__ Wm,   // [12288, 12288] row-major (K x N)
    const float* __restrict__ bias, // [12288]
    float* __restrict__ C)          // [64, 12288]
{
    const int lane = threadIdx.x & 31;
    const int wave = threadIdx.x >> 5;      // 0..7
    const int mt   = wave & 3;              // M tile index (0..3)
    const int np   = wave >> 2;             // N pair index within block (0..1)
    const int m0   = mt * 16;
    const int n0   = blockIdx.x * 64 + np * 32;   // wave covers cols n0..n0+31

    const int half = lane >> 4;             // 0: lanes 0-15, 1: lanes 16-31
    const int l15  = lane & 15;

    // A fragment addressing (contiguous K order, matching B):
    //   lanes 0-15 : row=l15, elems 0..15 -> K{0..15}
    //   lanes 16-31: row=l15, elems 0..15 -> K{16..31}
    const float* arow_ptr = A + (size_t)(m0 + l15) * NFEAT;
    const int akoff = half * 16;

    // B fragment addressing (contiguous K order, matching A):
    //   lanes 0-15 : col=l15, elems 0..15 -> K{0..15}
    //   lanes 16-31: col=l15, elems 0..15 -> K{16..31}
    const float* bcol0 = Wm + (size_t)(n0 + l15);        // first 16-col tile
    const float* bcol1 = Wm + (size_t)(n0 + 16 + l15);   // second 16-col tile
    const int bk0 = half * 16;

    v8f acc0 = {};
    v8f acc1 = {};

#pragma unroll 2
    for (int kc = 0; kc < NFEAT; kc += 32) {
        // ---- load + convert A fragment (one contiguous 16-float run) ----
        const float4 a0 = *reinterpret_cast<const float4*>(arow_ptr + kc + akoff);
        const float4 a1 = *reinterpret_cast<const float4*>(arow_ptr + kc + akoff + 4);
        const float4 a2 = *reinterpret_cast<const float4*>(arow_ptr + kc + akoff + 8);
        const float4 a3 = *reinterpret_cast<const float4*>(arow_ptr + kc + akoff + 12);
        v16bf af;
        af[0]  = f2bf(a0.x); af[1]  = f2bf(a0.y); af[2]  = f2bf(a0.z); af[3]  = f2bf(a0.w);
        af[4]  = f2bf(a1.x); af[5]  = f2bf(a1.y); af[6]  = f2bf(a1.z); af[7]  = f2bf(a1.w);
        af[8]  = f2bf(a2.x); af[9]  = f2bf(a2.y); af[10] = f2bf(a2.z); af[11] = f2bf(a2.w);
        af[12] = f2bf(a3.x); af[13] = f2bf(a3.y); af[14] = f2bf(a3.z); af[15] = f2bf(a3.w);

        // ---- load + convert both B fragments (64B-coalesced per half-wave) ----
        v16bf bf0, bf1;
#pragma unroll
        for (int i = 0; i < 16; ++i) {
            const size_t rowoff = (size_t)(kc + bk0 + i) * NFEAT;
            bf0[i] = f2bf(bcol0[rowoff]);
            bf1[i] = f2bf(bcol1[rowoff]);
        }

        // prefetch next K-slab of B (global_prefetch_b8)
        if (kc + 32 < NFEAT) {
            const size_t nxt = (size_t)(kc + 32 + bk0) * NFEAT;
            __builtin_prefetch(bcol0 + nxt, 0, 1);
            __builtin_prefetch(bcol1 + nxt, 0, 1);
        }

        // D = A x B + C   (v_wmma_f32_16x16x32_bf16), A shared by both tiles
        acc0 = __builtin_amdgcn_wmma_f32_16x16x32_bf16(
            false, af, false, bf0, (short)0, acc0, false, false);
        acc1 = __builtin_amdgcn_wmma_f32_16x16x32_bf16(
            false, af, false, bf1, (short)0, acc1, false, false);
    }

    // C layout: VGPR r -> row m0 + 8*half + r; col = n0 (+16) + l15
    const float bv0 = bias[n0 + l15];
    const float bv1 = bias[n0 + 16 + l15];
#pragma unroll
    for (int r = 0; r < 8; ++r) {
        const int row = m0 + half * 8 + r;
        C[(size_t)row * NFEAT + n0 + l15]      = acc0[r] + bv0;
        C[(size_t)row * NFEAT + n0 + 16 + l15] = acc1[r] + bv1;
    }
}

// ---------------------------------------------------------------------------
// Kernel 2: p[b, :] = softmax(phi[b, :, 2]) over H*W   (one block per batch)
// ---------------------------------------------------------------------------
__global__ __launch_bounds__(256) void softmax_kernel(
    const float* __restrict__ phi, float* __restrict__ p)
{
    __shared__ float red[256];
    const int b = blockIdx.x;
    const int t = threadIdx.x;
    const float* src = phi + (size_t)b * NFEAT;

    float m = -INFINITY;
    for (int i = t; i < HW; i += 256) m = fmaxf(m, src[3 * i + 2]);
    red[t] = m; __syncthreads();
    for (int s = 128; s > 0; s >>= 1) {
        if (t < s) red[t] = fmaxf(red[t], red[t + s]);
        __syncthreads();
    }
    const float mx = red[0];
    __syncthreads();

    float sum = 0.f;
    for (int i = t; i < HW; i += 256) sum += __expf(src[3 * i + 2] - mx);
    red[t] = sum; __syncthreads();
    for (int s = 128; s > 0; s >>= 1) {
        if (t < s) red[t] += red[t + s];
        __syncthreads();
    }
    const float inv = 1.0f / red[0];

    float* dst = p + (size_t)b * HW;
    for (int i = t; i < HW; i += 256) dst[i] = __expf(src[3 * i + 2] - mx) * inv;
}

// ---------------------------------------------------------------------------
// Kernel 3: K=64 Bellman sweeps, fully LDS-resident (one block per batch)
//   nv_d = p*V_shift_d + rin_shift_d - rout*(rin_shift_d != 0)
//   V    = max(V, max_d nv_d)
// LDS: Vpad(66x66) + RINpad(66x66) + rout(4096) + p(4096) = ~67.6 KB / WGP
// (320 KB available on CDNA5, so fine; zero HBM traffic inside the K loop)
// ---------------------------------------------------------------------------
__global__ __launch_bounds__(256) void vin_kernel(
    const float* __restrict__ phi, const float* __restrict__ p,
    float* __restrict__ Vout)
{
    extern __shared__ float smem[];
    float* Vpad = smem;                       // PS*PS
    float* Rpad = smem + PS * PS;             // PS*PS
    float* Rout = smem + 2 * PS * PS;         // HW
    float* Pm   = smem + 2 * PS * PS + HW;    // HW

    const int b = blockIdx.x;
    const int t = threadIdx.x;
    const float* phib = phi + (size_t)b * NFEAT;
    const float* pb   = p + (size_t)b * HW;

    for (int i = t; i < PS * PS; i += 256) { Vpad[i] = 0.f; Rpad[i] = 0.f; }
    __syncthreads();
    for (int i = t; i < HW; i += 256) {
        const int r = i >> 6, c = i & 63;
        Rpad[(r + 1) * PS + c + 1] = phib[3 * i + 0];
        Rout[i] = phib[3 * i + 1];
        Pm[i]   = pb[i];
    }
    __syncthreads();

    float newv[16];
    for (int it = 0; it < KITER; ++it) {
#pragma unroll
        for (int c16 = 0; c16 < 16; ++c16) {
            const int idx = t + c16 * 256;
            const int i = idx >> 6, j = idx & 63;
            const int pc = (i + 1) * PS + (j + 1);
            const float ro = Rout[idx];
            const float pv = Pm[idx];
            float best = Vpad[pc];
            // up (i-1,j)
            { const float rn = Rpad[pc - PS], vn = Vpad[pc - PS];
              best = fmaxf(best, pv * vn + rn - (rn != 0.f ? ro : 0.f)); }
            // down (i+1,j)
            { const float rn = Rpad[pc + PS], vn = Vpad[pc + PS];
              best = fmaxf(best, pv * vn + rn - (rn != 0.f ? ro : 0.f)); }
            // left (i,j-1)
            { const float rn = Rpad[pc - 1], vn = Vpad[pc - 1];
              best = fmaxf(best, pv * vn + rn - (rn != 0.f ? ro : 0.f)); }
            // right (i,j+1)
            { const float rn = Rpad[pc + 1], vn = Vpad[pc + 1];
              best = fmaxf(best, pv * vn + rn - (rn != 0.f ? ro : 0.f)); }
            newv[c16] = best;
        }
        __syncthreads();
#pragma unroll
        for (int c16 = 0; c16 < 16; ++c16) {
            const int idx = t + c16 * 256;
            const int i = idx >> 6, j = idx & 63;
            Vpad[(i + 1) * PS + (j + 1)] = newv[c16];
        }
        __syncthreads();
    }

    for (int i = t; i < HW; i += 256) {
        const int r = i >> 6, c = i & 63;
        Vout[(size_t)b * HW + i] = Vpad[(r + 1) * PS + c + 1];
    }
}

// ---------------------------------------------------------------------------
// Kernel 4: argmax(one-hot) -> 3x3x4 patch -> 36-16 relu -> 16-4 head
// (one block per batch)
// ---------------------------------------------------------------------------
__global__ __launch_bounds__(256) void head_kernel(
    const float* __restrict__ obs, const float* __restrict__ V,
    const float* __restrict__ w1, const float* __restrict__ b1,
    const float* __restrict__ w2, const float* __restrict__ b2,
    float* __restrict__ out)
{
    __shared__ float rv[256];
    __shared__ int   ri[256];
    __shared__ float patch[36];
    __shared__ float hbuf[16];

    const int b = blockIdx.x;
    const int t = threadIdx.x;
    const float* ob = obs + (size_t)b * NFEAT;

    float best = -INFINITY; int bi = 0x7FFFFFFF;
    for (int i = t; i < HW; i += 256) {
        const float v = ob[3 * i + 1];
        if (v > best) { best = v; bi = i; }
    }
    rv[t] = best; ri[t] = bi; __syncthreads();
    for (int s = 128; s > 0; s >>= 1) {
        if (t < s) {
            if (rv[t + s] > rv[t] || (rv[t + s] == rv[t] && ri[t + s] < ri[t])) {
                rv[t] = rv[t + s]; ri[t] = ri[t + s];
            }
        }
        __syncthreads();
    }
    const int flat = ri[0];
    const int pi = flat >> 6, pj = flat & 63;

    if (t < 36) {
        const int di = t / 12, dj = (t % 12) >> 2, ch = t & 3;
        const int gi = pi - 1 + di, gj = pj - 1 + dj;
        float val = 0.f;
        if (gi >= 0 && gi < HH && gj >= 0 && gj < WWID) {
            const int cell = gi * WWID + gj;
            val = (ch < 3) ? ob[3 * cell + ch] : V[(size_t)b * HW + cell];
        }
        patch[t] = val;
    }
    __syncthreads();

    if (t < 16) {
        float acc = b1[t];
#pragma unroll
        for (int i = 0; i < 36; ++i) acc += patch[i] * w1[i * 16 + t];
        hbuf[t] = fmaxf(acc, 0.f);
    }
    __syncthreads();

    if (t < 4) {
        float acc = b2[t];
#pragma unroll
        for (int i = 0; i < 16; ++i) acc += hbuf[i] * w2[i * 4 + t];
        out[b * 4 + t] = acc;
    }
}

// ---------------------------------------------------------------------------
// Host-side launcher
// ---------------------------------------------------------------------------
extern "C" void kernel_launch(void* const* d_in, const int* in_sizes, int n_in,
                              void* d_out, int out_size, void* d_ws, size_t ws_size,
                              hipStream_t stream)
{
    (void)in_sizes; (void)n_in; (void)out_size; (void)ws_size;

    const float* obs  = (const float*)d_in[0];  // [64, 12288]
    const float* phiw = (const float*)d_in[1];  // [12288, 12288]
    const float* phib = (const float*)d_in[2];  // [12288]
    const float* w1   = (const float*)d_in[3];  // [36, 16]
    const float* b1   = (const float*)d_in[4];  // [16]
    const float* w2   = (const float*)d_in[5];  // [16, 4]
    const float* b2   = (const float*)d_in[6];  // [4]
    float* out = (float*)d_out;                 // [64, 4]

    // workspace: phi (3 MB) | p (1 MB) | V (1 MB)
    float* phi = (float*)d_ws;
    float* p   = phi + (size_t)BB * NFEAT;
    float* V   = p + (size_t)BB * HW;

    phi_gemm_kernel<<<NFEAT / 64, 256, 0, stream>>>(obs, phiw, phib, phi);
    softmax_kernel<<<BB, 256, 0, stream>>>(phi, p);

    const size_t shmem = (size_t)(2 * PS * PS + 2 * HW) * sizeof(float);
    vin_kernel<<<BB, 256, shmem, stream>>>(phi, p, V);

    head_kernel<<<BB, 256, 0, stream>>>(obs, V, w1, b1, w2, b2, out);
}
